// Dense_test_1layer_29214367547644
// MI455X (gfx1250) — compile-verified, matrix operand order checked by reading, not run on hardware
//
#include <hip/hip_runtime.h>
#include <cstdint>

#define B_  256
#define T_  250
#define D_  700
#define H_  512
#define O_  20
#define BR_ 2
#define M_  (B_*T_)     // 64000 rows (b*T + t)
#define N1_ (H_*BR_)    // 1024
#define KP_ 704         // K padded to multiple of 32
#define NK_ (KP_/32)    // 22 k-steps
#define NO_ 32          // padded classifier outputs

typedef __bf16 bf16;
typedef __attribute__((ext_vector_type(16))) __bf16 v16bf;
typedef __attribute__((ext_vector_type(8)))  __bf16 v8bf;
typedef __attribute__((ext_vector_type(4)))  __bf16 v4bf;
typedef __attribute__((ext_vector_type(8)))  float  v8f;
typedef __attribute__((ext_vector_type(4)))  float  v4f;
typedef __attribute__((ext_vector_type(4)))  unsigned int u32x4;
typedef __attribute__((ext_vector_type(8)))  int i32x8;
typedef __attribute__((ext_vector_type(4)))  int i32x4;

#if __has_builtin(__builtin_amdgcn_tensor_load_to_lds) && __has_builtin(__builtin_amdgcn_s_wait_tensorcnt)
#define HAVE_TDM 1
#else
#define HAVE_TDM 0
#endif

__device__ __forceinline__ v16bf cat8(v8bf lo, v8bf hi) {
  return __builtin_shufflevector(lo, hi, 0,1,2,3,4,5,6,7,8,9,10,11,12,13,14,15);
}

// ---------------- weight prep: w_eff = W * mask, bf16, K padded -------------
__global__ __launch_bounds__(256) void k_wconv(const float* __restrict__ W,
                                               const unsigned char* __restrict__ mask,
                                               bf16* __restrict__ wb) {
  int idx = blockIdx.x * 256 + threadIdx.x;
  if (idx >= N1_ * KP_) return;
  int n = idx / KP_, k = idx - n * KP_;
  float v = 0.0f;
  if (k < D_ && mask[n * D_ + k]) v = W[n * D_ + k];
  wb[idx] = (bf16)v;
}

__global__ __launch_bounds__(256) void k_w2conv(const float* __restrict__ W2,
                                                bf16* __restrict__ w2b) {
  int idx = blockIdx.x * 256 + threadIdx.x;
  if (idx >= NO_ * H_) return;
  int n = idx / H_, k = idx - n * H_;
  w2b[idx] = (bf16)((n < O_) ? W2[n * H_ + k] : 0.0f);
}

__global__ void k_init(float* acc) {
  if (threadIdx.x == 0) { acc[0] = 0.0f; acc[1] = 0.0f; }
}

// ---------------- GEMM1: cur(bf16, Mx1024) = x(f32, Mx700) * wb^T ----------
// block tile 128x128, 8 waves (4M x 2N), wave tile 32x64 = 2x4 wmma tiles.
// LDS rows padded to 40 bf16 (80B) -> conflict-free ds_read_b128 fragments.
// B tile is DMA'd by the Tensor Data Mover (wave 0 issues, double-buffered,
// pad_enable generates the 80B row stride); A tile goes through VGPRs since
// it needs f32->bf16 conversion, with register prefetch overlapping WMMA.
__global__ __launch_bounds__(256) void k_gemm1(const float* __restrict__ x,
                                               const bf16* __restrict__ wb,
                                               bf16* __restrict__ cur) {
  __shared__ bf16 smA[128 * 40];
#if HAVE_TDM
  __shared__ bf16 smB[2 * 128 * 40];
#else
  __shared__ bf16 smB[128 * 40];
#endif
  const int tid = threadIdx.x;
  const int wave = tid >> 5, lane = tid & 31;
  const int l16 = lane & 15, lhi = lane >> 4;
  const int wm = wave & 3, wn = wave >> 2;
  const int m0 = blockIdx.x * 128, n0 = blockIdx.y * 128;

  v8f C[2][4] = {};
  v4f pa[4];

  auto loadA_full = [&](int kk) {               // 128 rows x 8 float4 chunks
    #pragma unroll
    for (int i = 0; i < 4; ++i) {
      int id = tid + i * 256;
      int row = id >> 3, kc = id & 7;
      pa[i] = *(const v4f*)(x + (size_t)(m0 + row) * D_ + kk * 32 + kc * 4);
    }
  };
  auto loadA_tail = [&]() {                     // kk = 21: k 672..699 valid
    #pragma unroll
    for (int i = 0; i < 4; ++i) {
      int id = tid + i * 256;
      int row = id >> 3, kc = id & 7;
      if (kc == 7) { v4f z = {0.f, 0.f, 0.f, 0.f}; pa[i] = z; }
      else pa[i] = *(const v4f*)(x + (size_t)(m0 + row) * D_ + 672 + kc * 4);
    }
  };
  auto stageA = [&]() {
    #pragma unroll
    for (int i = 0; i < 4; ++i) {
      int id = tid + i * 256;
      int row = id >> 3, kc = id & 7;
      v4bf t;
      t[0] = (bf16)pa[i][0]; t[1] = (bf16)pa[i][1];
      t[2] = (bf16)pa[i][2]; t[3] = (bf16)pa[i][3];
      *(v4bf*)(&smA[row * 40 + kc * 4]) = t;
    }
  };

#if HAVE_TDM
  // ---- Tensor DMA descriptor for one 128x32-bf16 B tile (uniform) ----
  const bool isW0 = (tid < 32);
  u32x4 g0;
  i32x8 g1;
  i32x4 gz = {0, 0, 0, 0};
  i32x8 gz8 = {0, 0, 0, 0, 0, 0, 0, 0};
  (void)gz8;
  const unsigned ldsB0 = (unsigned)(uintptr_t)(&smB[0]);   // low32 = LDS offset
  const uint64_t gaddr0 = (uint64_t)(uintptr_t)(wb + (size_t)n0 * KP_);
  // data_size=2B | pad_enable | pad_interval=16 DW (64B) | pad_amount=4 DW (16B)
  g1[0] = 0x06D10000;
  g1[1] = (KP_ << 16);          // tensor_dim0 = 704
  g1[2] = (N1_ << 16);          // tensor_dim1 = 1024
  g1[3] = (32 << 16);           // tile_dim0 = 32
  g1[4] = 128;                  // tile_dim1 = 128, tile_dim2 = 0
  g1[5] = KP_;                  // tensor_dim0_stride = 704
  g1[6] = 0; g1[7] = 0;
  auto tdmB = [&](int kk, int buf) {
    uint64_t ga = gaddr0 + (uint64_t)kk * 64;  // kk*32 elems * 2B
    g0[0] = 1;                                 // count=1, user mode
    g0[1] = ldsB0 + buf * (128 * 40 * 2);      // lds_addr
    g0[2] = (unsigned)ga;                      // global_addr[31:0]
    g0[3] = (unsigned)(ga >> 32) | 0x80000000u;// global_addr[56:32] | type=2
#if defined(__clang_major__) && __clang_major__ >= 23
    __builtin_amdgcn_tensor_load_to_lds(g0, g1, gz, gz, gz8, 0);
#else
    __builtin_amdgcn_tensor_load_to_lds(g0, g1, gz, gz, 0);
#endif
  };

  if (isW0) tdmB(0, 0);
  loadA_full(0);
  for (int kk = 0; kk < NK_; ++kk) {
    __syncthreads();                           // prev compute done
    stageA();
    if (isW0) {
      if (kk + 1 < NK_) {
        tdmB(kk + 1, (kk + 1) & 1);            // prefetch next B tile
        __builtin_amdgcn_s_wait_tensorcnt(1);  // tile kk complete, kk+1 in flight
      } else {
        __builtin_amdgcn_s_wait_tensorcnt(0);
      }
    }
    __syncthreads();                           // smA + smB[kk&1] ready for all
    if (kk + 1 < NK_) {                        // overlap global A with wmma
      if (kk + 1 == NK_ - 1) loadA_tail(); else loadA_full(kk + 1);
    }
    const bf16* smBcur = &smB[(kk & 1) * (128 * 40)];
#else
  v8bf pb[2];
  auto loadB = [&](int kk) {                   // 128 rows x 4 b128 chunks
    #pragma unroll
    for (int i = 0; i < 2; ++i) {
      int id = tid + i * 256;
      int row = id >> 2, kc = id & 3;
      pb[i] = *(const v8bf*)(wb + (size_t)(n0 + row) * KP_ + kk * 32 + kc * 8);
    }
  };
  auto stageB = [&]() {
    #pragma unroll
    for (int i = 0; i < 2; ++i) {
      int id = tid + i * 256;
      int row = id >> 2, kc = id & 3;
      *(v8bf*)(&smB[row * 40 + kc * 8]) = pb[i];
    }
  };
  loadA_full(0); loadB(0);
  for (int kk = 0; kk < NK_; ++kk) {
    __syncthreads();
    stageA(); stageB();
    __syncthreads();
    if (kk + 1 < NK_) {
      if (kk + 1 == NK_ - 1) loadA_tail(); else loadA_full(kk + 1);
      loadB(kk + 1);
    }
    const bf16* smBcur = &smB[0];
#endif
    v16bf afr[2], bfr[4];
    #pragma unroll
    for (int mi = 0; mi < 2; ++mi) {
      const bf16* p = &smA[(wm * 32 + mi * 16 + l16) * 40 + lhi * 8];
      afr[mi] = cat8(*(const v8bf*)p, *(const v8bf*)(p + 16));
    }
    #pragma unroll
    for (int ni = 0; ni < 4; ++ni) {
      const bf16* p = &smBcur[(wn * 64 + ni * 16 + l16) * 40 + lhi * 8];
      bfr[ni] = cat8(*(const v8bf*)p, *(const v8bf*)(p + 16));
    }
    #pragma unroll
    for (int mi = 0; mi < 2; ++mi)
      #pragma unroll
      for (int ni = 0; ni < 4; ++ni)
        C[mi][ni] = __builtin_amdgcn_wmma_f32_16x16x32_bf16(
            false, afr[mi], false, bfr[ni], (short)0, C[mi][ni], false, false);
  }

  #pragma unroll
  for (int mi = 0; mi < 2; ++mi)
    #pragma unroll
    for (int ni = 0; ni < 4; ++ni) {
      int mbase = m0 + wm * 32 + mi * 16 + lhi * 8;
      int ncol  = n0 + wn * 64 + ni * 16 + l16;
      #pragma unroll
      for (int r = 0; r < 8; ++r)
        cur[(size_t)(mbase + r) * N1_ + ncol] = (bf16)C[mi][ni][r];
    }
}

// ---------------- temporal scan: LIF with dendritic branches ---------------
__global__ __launch_bounds__(256) void k_scan(const bf16* __restrict__ cur,
                                              const float* __restrict__ mem0,
                                              const float* __restrict__ tau_m,
                                              const float* __restrict__ tau_n,
                                              bf16* __restrict__ spk) {
  int gid = blockIdx.x * 256 + threadIdx.x;     // B*H threads
  int b = gid >> 9, h = gid & (H_ - 1);
  float alpha = 1.0f / (1.0f + __expf(-tau_m[h]));
  float beta0 = 1.0f / (1.0f + __expf(-tau_n[h * 2 + 0]));
  float beta1 = 1.0f / (1.0f + __expf(-tau_n[h * 2 + 1]));
  float mem = mem0[b * H_ + h], s = 0.0f, d0 = 0.0f, d1 = 0.0f;
  const unsigned* cp = (const unsigned*)cur;    // 2 packed bf16 branches
  for (int t = 0; t < T_; ++t) {
    unsigned v = cp[(size_t)(b * T_ + t) * (N1_ / 2) + h];
    float c0 = __uint_as_float(v << 16);
    float c1 = __uint_as_float(v & 0xffff0000u);
    d0 = beta0 * d0 + (1.0f - beta0) * c0;
    d1 = beta1 * d1 + (1.0f - beta1) * c1;
    float l = d0 + d1;
    mem = (mem - s) * alpha + (1.0f - alpha) * l;   // VTH = 1
    s = (mem > 1.0f) ? 1.0f : 0.0f;
    spk[(size_t)(t * B_ + b) * H_ + h] = (bf16)s;
  }
}

// ---------------- GEMM2: logits(f32, Mx32) = spk * w2b^T + b2 --------------
__global__ __launch_bounds__(256) void k_gemm2(const bf16* __restrict__ spk,
                                               const bf16* __restrict__ w2b,
                                               const float* __restrict__ b2,
                                               float* __restrict__ logits) {
  int tid = threadIdx.x, wave = tid >> 5, lane = tid & 31;
  int l16 = lane & 15, lhi = lane >> 4;
  int arow = blockIdx.x * 128 + wave * 16 + l16;
  const bf16* pa = spk + (size_t)arow * H_;
  v8f C[2] = {};
  for (int k0 = 0; k0 < H_; k0 += 32) {
    v16bf a = cat8(*(const v8bf*)(pa + k0 + lhi * 8),
                   *(const v8bf*)(pa + k0 + 16 + lhi * 8));
    #pragma unroll
    for (int ni = 0; ni < 2; ++ni) {
      const bf16* pb = w2b + (size_t)(ni * 16 + l16) * H_ + k0 + lhi * 8;
      v16bf bb = cat8(*(const v8bf*)pb, *(const v8bf*)(pb + 16));
      C[ni] = __builtin_amdgcn_wmma_f32_16x16x32_bf16(
          false, a, false, bb, (short)0, C[ni], false, false);
    }
  }
  #pragma unroll
  for (int ni = 0; ni < 2; ++ni) {
    int ncol = ni * 16 + l16;
    float bias = (ncol < O_) ? b2[ncol] : 0.0f;
    int mbase = blockIdx.x * 128 + wave * 16 + lhi * 8;
    #pragma unroll
    for (int r = 0; r < 8; ++r)
      logits[(size_t)(mbase + r) * NO_ + ncol] = C[ni][r] + bias;
  }
}

// ---------------- per-row logsumexp / NLL / argmax + reduction -------------
__global__ __launch_bounds__(256) void k_loss(const float* __restrict__ logits,
                                              const int* __restrict__ target,
                                              float* __restrict__ acc) {
  __shared__ float snll[256];
  __shared__ float scor[256];
  int r = blockIdx.x * 256 + threadIdx.x;       // row id = t*B + b
  int t = r >> 8, b = r & 255;
  const float* lp = logits + (size_t)r * NO_;
  float l[O_];
  #pragma unroll
  for (int i = 0; i < 5; ++i) {
    v4f v = *(const v4f*)(lp + i * 4);
    l[i*4+0] = v[0]; l[i*4+1] = v[1]; l[i*4+2] = v[2]; l[i*4+3] = v[3];
  }
  float m = l[0]; int am = 0;
  #pragma unroll
  for (int o = 1; o < O_; ++o) if (l[o] > m) { m = l[o]; am = o; }
  float se = 0.0f;
  #pragma unroll
  for (int o = 0; o < O_; ++o) se += __expf(l[o] - m);
  float lse = m + __logf(se);
  int tg = target[b * T_ + t];
  snll[threadIdx.x] = lse - l[tg];
  scor[threadIdx.x] = (am == tg) ? 1.0f : 0.0f;
  __syncthreads();
  for (int s = 128; s > 0; s >>= 1) {
    if (threadIdx.x < s) {
      snll[threadIdx.x] += snll[threadIdx.x + s];
      scor[threadIdx.x] += scor[threadIdx.x + s];
    }
    __syncthreads();
  }
  if (threadIdx.x == 0) { atomicAdd(&acc[0], snll[0]); atomicAdd(&acc[1], scor[0]); }
}

__global__ void k_final(const float* __restrict__ acc, float* __restrict__ out) {
  out[0] = acc[0] / (float)M_;   // mean NLL over T*B
  out[1] = acc[1];               // correct count
  out[2] = (float)M_;            // total
}

// ---------------------------------------------------------------------------
extern "C" void kernel_launch(void* const* d_in, const int* in_sizes, int n_in,
                              void* d_out, int out_size, void* d_ws, size_t ws_size,
                              hipStream_t stream) {
  const float* x      = (const float*)d_in[0];
  const int*   target = (const int*)d_in[1];
  const float* mem0   = (const float*)d_in[2];
  const float* W      = (const float*)d_in[3];
  const float* tau_m  = (const float*)d_in[4];
  const float* tau_n  = (const float*)d_in[5];
  const float* W2     = (const float*)d_in[6];
  const float* b2     = (const float*)d_in[7];
  const unsigned char* mask = (const unsigned char*)d_in[8];
  float* out = (float*)d_out;

  char* base = (char*)d_ws;
  size_t off = 0;
  auto alloc = [&](size_t bytes) -> char* {
    char* p = base + off;
    off = (off + bytes + 255) & ~(size_t)255;
    return p;
  };
  bf16*  wb     = (bf16*)alloc((size_t)N1_ * KP_ * 2);   //  1.4 MB
  bf16*  w2b    = (bf16*)alloc((size_t)NO_ * H_ * 2);    //  32 KB
  bf16*  cur    = (bf16*)alloc((size_t)M_ * N1_ * 2);    // 131 MB
  bf16*  spk    = (bf16*)alloc((size_t)M_ * H_ * 2);     //  65 MB
  float* logits = (float*)alloc((size_t)M_ * NO_ * 4);   //   8 MB
  float* acc    = (float*)alloc(256);
  (void)in_sizes; (void)n_in; (void)out_size; (void)ws_size;

  k_init  <<<1, 64, 0, stream>>>(acc);
  k_wconv <<<(N1_ * KP_ + 255) / 256, 256, 0, stream>>>(W, mask, wb);
  k_w2conv<<<(NO_ * H_ + 255) / 256, 256, 0, stream>>>(W2, w2b);
  k_gemm1 <<<dim3(M_ / 128, N1_ / 128), 256, 0, stream>>>(x, wb, cur);
  k_scan  <<<(B_ * H_) / 256, 256, 0, stream>>>(cur, mem0, tau_m, tau_n, spk);
  k_gemm2 <<<M_ / 128, 256, 0, stream>>>(spk, w2b, b2, logits);
  k_loss  <<<M_ / 256, 256, 0, stream>>>(logits, target, acc);
  k_final <<<1, 1, 0, stream>>>(acc, out);
}
